// RNN_35287451304968
// MI455X (gfx1250) — compile-verified, matrix operand order checked by reading, not run on hardware
//
#include <hip/hip_runtime.h>
#include <hip/hip_bf16.h>
#include <cstdint>
#include <cstddef>

typedef __attribute__((ext_vector_type(16))) __bf16 v16bf;
typedef __attribute__((ext_vector_type(8)))  float  v8f;
typedef __attribute__((ext_vector_type(4)))  unsigned int ui32x4;
typedef __attribute__((ext_vector_type(8)))  int   i32x8;
typedef __attribute__((ext_vector_type(4)))  int   i32x4;
typedef unsigned short u16;

constexpr int kB = 4096, kT = 512, kD = 13, kH = 64, kC = 11;
constexpr float kEps = 1e-5f;

// ---- workspace layout ----
// xa: packed layernormed input as bf16 A-fragments: [B/16 tiles][T][32 lanes][16 elems]
constexpr size_t kXAElems = (size_t)(kB / 16) * kT * 512;   // u16 elements
constexpr size_t kXABytes = kXAElems * sizeof(u16);         // 134 MB
// cflat: [B][128] f32 (c0 | c1)  -> 2 MB, placed after xa

// ---- LDS layout (u16 units) ----
constexpr int OFF_WIH0 = 0;        // 16 tiles * 512           =  8192 (16 KB)
constexpr int OFF_WHH0 = 8192;     // 16 tiles * 2 frag * 512  = 16384 (32 KB)
constexpr int OFF_WIH1 = 24576;    // 16384 (32 KB)
constexpr int OFF_WHH1 = 40960;    // 16384 (32 KB)
constexpr int OFF_H    = 57344;    // 4 waves * 4096 (2 layers * 2 bufs * 1024) (32 KB)
constexpr int OFF_XS   = 73728;    // 4 waves * 2 bufs * 512 u16 = 4096 (8 KB) TDM staging
constexpr int kSmemU16   = 77824;
constexpr int kSmemBytes = kSmemU16 * 2;   // 155648 B of 320 KB WGP LDS

__device__ __forceinline__ u16 f2bf(float f) {
  unsigned u = __float_as_uint(f);
  u += 0x7fffu + ((u >> 16) & 1u);     // round-to-nearest-even
  return (u16)(u >> 16);
}
__device__ __forceinline__ float sigm(float x) {
  return 1.0f / (1.0f + __expf(-x));
}
__device__ __forceinline__ float tanh_fast(float x) {
  float xc = fminf(fmaxf(x, -15.0f), 15.0f);
  float e = __expf(-2.0f * xc);
  return (1.0f - e) / (1.0f + e);
}

#define WMMA_BF16(Av, Bv, Cv)                                                   \
  __builtin_amdgcn_wmma_f32_16x16x32_bf16(false, (Av), false, (Bv), (short)0,   \
                                          (Cv), false, false)

// B-fragment loaders from LDS (32 bytes per lane, contiguous)
__device__ __forceinline__ v16bf ldB2(const u16* base, int nt, int frag, int lane) {
  return *(const v16bf*)(base + (((nt << 1) + frag) << 9) + (lane << 4));
}
__device__ __forceinline__ v16bf ldB1(const u16* base, int nt, int lane) {
  return *(const v16bf*)(base + (nt << 9) + (lane << 4));
}

// Tensor Data Mover: 1 KB linear tile (128 x 8B), global -> LDS.
// D# per CDNA5 ISA 8.3/8.4: group0 = count|lds_addr|global_addr|type=2,
// group1 = data_size=8B, tensor_dim0 = tile_dim0 = 128, 1-D tile; groups 2/3 zero.
// This toolchain's builtin is the 6-arg form:
//   (uint32x4 g0, int32x8 g1, int32x4 g2, int32x4 g3, int32x8 pad, i32 cpol)
__device__ __forceinline__ void tdm_load_x(unsigned lds_byte_off, const u16* gsrc) {
  unsigned long long ga = (unsigned long long)(uintptr_t)gsrc;
  ui32x4 g0;
  g0[0] = 1u;                                            // count = 1 (valid user D#)
  g0[1] = lds_byte_off;                                  // lds_addr
  g0[2] = (unsigned)(ga & 0xffffffffu);                  // global_addr[31:0]
  g0[3] = (2u << 30) | (unsigned)((ga >> 32) & 0x01ffffffu); // type=2 | addr[56:32]
  i32x8 g1;
  g1[0] = (3 << 16);                 // workgroup_mask=0, data_size=3 (8 bytes)
  g1[1] = (int)(128u << 16);         // tensor_dim0[15:0] = 128 in bits [63:48]
  g1[2] = (1 << 16);                 // tensor_dim1[15:0] = 1 in bits [95:80]
  g1[3] = (int)(128u << 16);         // tile_dim0 = 128 in bits [127:112]
  g1[4] = 0;                         // tile_dim1 = 0 (unused), tile_dim2 = 0
  g1[5] = 128;                       // tensor_dim0_stride = 128
  g1[6] = 0;
  g1[7] = 0;
  i32x4 z4 = {0, 0, 0, 0};
  i32x8 z8 = {0, 0, 0, 0, 0, 0, 0, 0};
  __builtin_amdgcn_tensor_load_to_lds(g0, g1, z4, z4, z8, 0);
}

// ---------------------------------------------------------------------------
// Kernel 1: input layernorm + pack into bf16 WMMA A-fragment layout (K pad 13->32)
// A-layout (16-bit 16x32): lane = M + (inner8 ? 16 : 0); within a lane,
// element j holds K = 16*(j>>3) + (lane>=16 ? 8 : 0) + (j&7).
// ---------------------------------------------------------------------------
__global__ void ln_pack_kernel(const float* __restrict__ x,
                               const float* __restrict__ g,
                               const float* __restrict__ b,
                               u16* __restrict__ xa) {
  int idx = blockIdx.x * blockDim.x + threadIdx.x;
  if (idx >= kB * kT) return;
  int bi = idx / kT;
  int t  = idx - bi * kT;
  const float* xp = x + (size_t)idx * kD;

  float xv[kD];
  float s = 0.f;
#pragma unroll
  for (int k = 0; k < kD; ++k) { xv[k] = xp[k]; s += xv[k]; }
  float m = s * (1.0f / kD);
  float v = 0.f;
#pragma unroll
  for (int k = 0; k < kD; ++k) { float d = xv[k] - m; v += d * d; }
  v *= (1.0f / kD);
  float r = rsqrtf(v + kEps);

  u16 lo[16], hi[16];
#pragma unroll
  for (int k = 0; k < 16; ++k) { lo[k] = 0; hi[k] = 0; }
#pragma unroll
  for (int k = 0; k < 8; ++k)  lo[k]     = f2bf((xv[k] - m) * r * g[k] + b[k]); // K=0..7
#pragma unroll
  for (int k = 8; k < kD; ++k) hi[k - 8] = f2bf((xv[k] - m) * r * g[k] + b[k]); // K=8..12

  // row r of tile -> lane r (K 0..7,16..23) and lane r+16 (K 8..15,24..31); K>=13 zero
  u16* dst = xa + (((size_t)(bi >> 4) * kT + t) << 9) + ((size_t)(bi & 15) << 4);
#pragma unroll
  for (int k = 0; k < 16; ++k) { dst[k] = lo[k]; dst[256 + k] = hi[k]; }
}

// ---------------------------------------------------------------------------
// Kernel 2: fused 2-layer LSTM scan. 4 waves/block, each wave owns 16 batch
// rows for both layers over all T steps. Weights in LDS as bf16 B-fragments.
// h double-buffered in LDS (per wave, per layer); c lives in VGPRs.
// x[t] streamed via TDM into a per-wave double-buffered LDS stage, prefetched
// one step ahead (tensorcnt <= 1 keeps the next-step DMA in flight).
// ---------------------------------------------------------------------------
__global__ __launch_bounds__(128, 1) void lstm_kernel(
    const u16*  __restrict__ xa,
    const float* __restrict__ wih0, const float* __restrict__ whh0,
    const float* __restrict__ wih1, const float* __restrict__ whh1,
    float* __restrict__ cflat) {
  extern __shared__ __align__(32) char smem_raw[];
  u16* smem   = (u16*)smem_raw;
  u16* s_wih0 = smem + OFF_WIH0;
  u16* s_whh0 = smem + OFF_WHH0;
  u16* s_wih1 = smem + OFF_WIH1;
  u16* s_whh1 = smem + OFF_WHH1;
  u16* s_h    = smem + OFF_H;
  u16* s_x    = smem + OFF_XS;

  // ---- cooperative weight pack (fp32 -> bf16, B-fragment layout) ----
  // B[k][n] = W[gate = 16*nt + (lane&15)][k],   k = 32*frag + k'(lane, j)
  for (int e = threadIdx.x; e < 8192; e += 128) {        // W_ih0 [256 x 13], K pad 32
    int nt    = e >> 9;
    int lanei = (e >> 4) & 31;
    int j     = e & 15;
    int N = (nt << 4) + (lanei & 15);
    int k = ((j >> 3) << 4) + ((lanei & 16) ? 8 : 0) + (j & 7);
    s_wih0[e] = (k < kD) ? f2bf(wih0[N * kD + k]) : (u16)0;
  }
  for (int e = threadIdx.x; e < 16384; e += 128) {       // three [256 x 64] weights
    int nt    = e >> 10;
    int frag  = (e >> 9) & 1;
    int lanei = (e >> 4) & 31;
    int j     = e & 15;
    int N = (nt << 4) + (lanei & 15);
    int k = (frag << 5) + ((j >> 3) << 4) + ((lanei & 16) ? 8 : 0) + (j & 7);
    s_whh0[e] = f2bf(whh0[N * kH + k]);
    s_wih1[e] = f2bf(wih1[N * kH + k]);
    s_whh1[e] = f2bf(whh1[N * kH + k]);
  }
  __syncthreads();

  const int w    = threadIdx.x >> 5;
  const int lane = threadIdx.x & 31;
  const int hl   = lane & 15;     // N / hidden-within-chunk index of C/D layout
  const int mh   = lane >> 4;     // row half of C/D layout

  u16* hb = s_h + (w << 12);      // this wave's 4096-u16 h region
  for (int i = lane; i < 1024; i += 32) { hb[i] = 0; hb[2048 + i] = 0; } // zero buf0

  const int tile = blockIdx.x * 4 + w;               // 16-row batch tile id
  const u16* xaP = xa + (size_t)tile * kT * 512;

  // per-wave x staging: 2 buffers x 512 u16 (1 KB each)
  u16* xs = s_x + (w << 10);
  const unsigned xsByteBase = (unsigned)((OFF_XS << 1) + (w << 11));
  tdm_load_x(xsByteBase, xaP);                       // prefetch t = 0 into buf 0

  float c0[4][8], c1[4][8];
#pragma unroll
  for (int n = 0; n < 4; ++n)
#pragma unroll
    for (int r = 0; r < 8; ++r) { c0[n][r] = 0.f; c1[n][r] = 0.f; }

  for (int t = 0; t < kT; ++t) {
    // prefetch next step (last step prefetches a harmless dummy tile)
    const int tn = (t + 1 < kT) ? (t + 1) : 0;
    tdm_load_x(xsByteBase + (unsigned)((tn & 1) << 10), xaP + ((size_t)tn << 9));
    __builtin_amdgcn_s_wait_tensorcnt(1);            // current buffer ready, next in flight

    const int rb = t & 1, wbf = rb ^ 1;
    const u16* h0r = hb + (rb << 10);
    u16*       h0w = hb + (wbf << 10);
    const u16* h1r = hb + 2048 + (rb << 10);
    u16*       h1w = hb + 2048 + (wbf << 10);

    v16bf xf  = *(const v16bf*)(xs + ((t & 1) << 9) + (lane << 4));
    v16bf a00 = *(const v16bf*)(h0r + (lane << 4));
    v16bf a01 = *(const v16bf*)(h0r + 512 + (lane << 4));

    // ---------------- layer 0 ----------------
#pragma unroll
    for (int n = 0; n < 4; ++n) {                 // hidden chunk: gates at nt = n, n+4, n+8, n+12
      v8f aI = {}, aF = {}, aG = {}, aO = {};
      aI = WMMA_BF16(xf, ldB1(s_wih0, n,      lane), aI);
      aF = WMMA_BF16(xf, ldB1(s_wih0, n + 4,  lane), aF);
      aG = WMMA_BF16(xf, ldB1(s_wih0, n + 8,  lane), aG);
      aO = WMMA_BF16(xf, ldB1(s_wih0, n + 12, lane), aO);
      aI = WMMA_BF16(a00, ldB2(s_whh0, n,      0, lane), aI);
      aI = WMMA_BF16(a01, ldB2(s_whh0, n,      1, lane), aI);
      aF = WMMA_BF16(a00, ldB2(s_whh0, n + 4,  0, lane), aF);
      aF = WMMA_BF16(a01, ldB2(s_whh0, n + 4,  1, lane), aF);
      aG = WMMA_BF16(a00, ldB2(s_whh0, n + 8,  0, lane), aG);
      aG = WMMA_BF16(a01, ldB2(s_whh0, n + 8,  1, lane), aG);
      aO = WMMA_BF16(a00, ldB2(s_whh0, n + 12, 0, lane), aO);
      aO = WMMA_BF16(a01, ldB2(s_whh0, n + 12, 1, lane), aO);
#pragma unroll
      for (int r = 0; r < 8; ++r) {
        float iv = sigm(aI[r]), fv = sigm(aF[r]);
        float gv = tanh_fast(aG[r]), ov = sigm(aO[r]);
        float c = fv * c0[n][r] + iv * gv;
        c0[n][r] = c;
        float hv = ov * tanh_fast(c);
        // scatter h into A-fragment layout of the write buffer
        int k  = (n << 4) + hl;           // hidden index 0..63
        int fr = k >> 5, kk = k & 31;
        int dl = (r + (mh << 3)) + ((kk & 8) ? 16 : 0);      // dest lane
        int jj = ((kk >> 4) << 3) + (kk & 7);                // dest elem
        h0w[(fr << 9) + (dl << 4) + jj] = f2bf(hv);
      }
    }

    // ---------------- layer 1 (input = freshly written h0) ----------------
    v16bf b00 = *(const v16bf*)(h0w + (lane << 4));
    v16bf b01 = *(const v16bf*)(h0w + 512 + (lane << 4));
    v16bf a10 = *(const v16bf*)(h1r + (lane << 4));
    v16bf a11 = *(const v16bf*)(h1r + 512 + (lane << 4));

#pragma unroll
    for (int n = 0; n < 4; ++n) {
      v8f aI = {}, aF = {}, aG = {}, aO = {};
      aI = WMMA_BF16(b00, ldB2(s_wih1, n,      0, lane), aI);
      aI = WMMA_BF16(b01, ldB2(s_wih1, n,      1, lane), aI);
      aF = WMMA_BF16(b00, ldB2(s_wih1, n + 4,  0, lane), aF);
      aF = WMMA_BF16(b01, ldB2(s_wih1, n + 4,  1, lane), aF);
      aG = WMMA_BF16(b00, ldB2(s_wih1, n + 8,  0, lane), aG);
      aG = WMMA_BF16(b01, ldB2(s_wih1, n + 8,  1, lane), aG);
      aO = WMMA_BF16(b00, ldB2(s_wih1, n + 12, 0, lane), aO);
      aO = WMMA_BF16(b01, ldB2(s_wih1, n + 12, 1, lane), aO);
      aI = WMMA_BF16(a10, ldB2(s_whh1, n,      0, lane), aI);
      aI = WMMA_BF16(a11, ldB2(s_whh1, n,      1, lane), aI);
      aF = WMMA_BF16(a10, ldB2(s_whh1, n + 4,  0, lane), aF);
      aF = WMMA_BF16(a11, ldB2(s_whh1, n + 4,  1, lane), aF);
      aG = WMMA_BF16(a10, ldB2(s_whh1, n + 8,  0, lane), aG);
      aG = WMMA_BF16(a11, ldB2(s_whh1, n + 8,  1, lane), aG);
      aO = WMMA_BF16(a10, ldB2(s_whh1, n + 12, 0, lane), aO);
      aO = WMMA_BF16(a11, ldB2(s_whh1, n + 12, 1, lane), aO);
#pragma unroll
      for (int r = 0; r < 8; ++r) {
        float iv = sigm(aI[r]), fv = sigm(aF[r]);
        float gv = tanh_fast(aG[r]), ov = sigm(aO[r]);
        float c = fv * c1[n][r] + iv * gv;
        c1[n][r] = c;
        float hv = ov * tanh_fast(c);
        int k  = (n << 4) + hl;
        int fr = k >> 5, kk = k & 31;
        int dl = (r + (mh << 3)) + ((kk & 8) ? 16 : 0);
        int jj = ((kk >> 4) << 3) + (kk & 7);
        h1w[(fr << 9) + (dl << 4) + jj] = f2bf(hv);
      }
    }
  }

  // drain the trailing dummy prefetch before the wave exits
  __builtin_amdgcn_s_wait_tensorcnt(0);

  // ---- store final cell states: cflat[row][0:64]=c0, [64:128]=c1 ----
  const int rowBase = tile << 4;
#pragma unroll
  for (int n = 0; n < 4; ++n)
#pragma unroll
    for (int r = 0; r < 8; ++r) {
      int row = rowBase + r + (mh << 3);
      cflat[(size_t)row * 128 + (n << 4) + hl]      = c0[n][r];
      cflat[(size_t)row * 128 + 64 + (n << 4) + hl] = c1[n][r];
    }
}

// ---------------------------------------------------------------------------
// Kernel 3: layernorm(128) + [128 -> 11] linear + softmax, one thread per row
// ---------------------------------------------------------------------------
__global__ void head_kernel(const float* __restrict__ cflat,
                            const float* __restrict__ g,
                            const float* __restrict__ b,
                            const float* __restrict__ Wo,
                            const float* __restrict__ bo,
                            float* __restrict__ out) {
  int row = blockIdx.x * blockDim.x + threadIdx.x;
  if (row >= kB) return;
  const float* cr = cflat + (size_t)row * 128;

  float s = 0.f;
  for (int k = 0; k < 128; ++k) s += cr[k];
  float m = s * (1.0f / 128.0f);
  float v = 0.f;
  for (int k = 0; k < 128; ++k) { float d = cr[k] - m; v += d * d; }
  v *= (1.0f / 128.0f);
  float r = rsqrtf(v + kEps);

  float acc[kC];
#pragma unroll
  for (int c = 0; c < kC; ++c) acc[c] = bo[c];
  for (int k = 0; k < 128; ++k) {
    float hn = (cr[k] - m) * r * g[k] + b[k];
#pragma unroll
    for (int c = 0; c < kC; ++c) acc[c] += hn * Wo[c * 128 + k];
  }

  float mx = acc[0];
#pragma unroll
  for (int c = 1; c < kC; ++c) mx = fmaxf(mx, acc[c]);
  float se = 0.f;
#pragma unroll
  for (int c = 0; c < kC; ++c) { float e = __expf(acc[c] - mx); acc[c] = e; se += e; }
  float inv = 1.0f / se;
#pragma unroll
  for (int c = 0; c < kC; ++c) out[(size_t)row * kC + c] = acc[c] * inv;
}

// ---------------------------------------------------------------------------
extern "C" void kernel_launch(void* const* d_in, const int* in_sizes, int n_in,
                              void* d_out, int out_size, void* d_ws, size_t ws_size,
                              hipStream_t stream) {
  const float* x    = (const float*)d_in[0];
  const float* lng  = (const float*)d_in[1];
  const float* lnb  = (const float*)d_in[2];
  const float* wih0 = (const float*)d_in[3];
  const float* whh0 = (const float*)d_in[4];
  const float* wih1 = (const float*)d_in[5];
  const float* whh1 = (const float*)d_in[6];
  const float* lncg = (const float*)d_in[7];
  const float* lncb = (const float*)d_in[8];
  const float* wout = (const float*)d_in[9];
  const float* bout = (const float*)d_in[10];
  float* out = (float*)d_out;

  u16*   xa    = (u16*)d_ws;
  float* cflat = (float*)((char*)d_ws + kXABytes);

  (void)in_sizes; (void)n_in; (void)out_size; (void)ws_size;

  (void)hipFuncSetAttribute(reinterpret_cast<const void*>(lstm_kernel),
                            hipFuncAttributeMaxDynamicSharedMemorySize, kSmemBytes);

  int nln = kB * kT;
  hipLaunchKernelGGL(ln_pack_kernel, dim3((nln + 255) / 256), dim3(256), 0, stream,
                     x, lng, lnb, xa);
  hipLaunchKernelGGL(lstm_kernel, dim3(kB / 64), dim3(128), kSmemBytes, stream,
                     xa, wih0, whh0, wih1, whh1, cflat);
  hipLaunchKernelGGL(head_kernel, dim3((kB + 255) / 256), dim3(256), 0, stream,
                     cflat, lncg, lncb, wout, bout, out);
}
